// TSPDecoder_4724464025925
// MI455X (gfx1250) — matrix operand order repaired and checked by reading, not compiled, not run on hardware
//
#include <hip/hip_runtime.h>
#include <hip/hip_fp16.h>
#include <stdint.h>

// ---------------------------------------------------------------------------
// Kool pointer-decoder for TSP on gfx1250 (MI455X).
//   B=2048, N=128, d=128, H=8, dk=16.
//   scores[b,n] = 10*tanh( (K[b,n,:]·q[b,:]) / 32 ),  q = qg + h_cur @ Wcq_bot
//   K  : one-time WMMA f16 GEMM (A tile staged via TDM tensor_load_to_lds),
//        stored packed-transposed Kt[b][d/4][node][4] (f16, 64MB, L2-resident)
//   q  : per-step WMMA f16 GEMM over gathered h_cur rows
//   GEMV+sample: coalesced 8B/lane f16 streaming + Gumbel-max categorical
// ---------------------------------------------------------------------------

typedef __attribute__((ext_vector_type(16))) _Float16 v16h;
typedef __attribute__((ext_vector_type(8)))  float    v8f;
typedef __attribute__((ext_vector_type(4)))  unsigned int v4u;
typedef __attribute__((ext_vector_type(8)))  int      v8i_;
typedef __attribute__((ext_vector_type(4)))  int      v4i_;

union Frag { v16h v; uint4 q[2]; };

#define NB   2048
#define NN   128
#define DD   128

__device__ __forceinline__ uint32_t hashu(uint32_t x) {
  x ^= x >> 16; x *= 0x7feb352du;
  x ^= x >> 15; x *= 0x846ca68bu;
  x ^= x >> 16; return x;
}

// ---------------------------------------------------------------------------
// One-time weight prep
// ---------------------------------------------------------------------------
__global__ void prep_wkT(const float* __restrict__ Wk, _Float16* __restrict__ WkT) {
  int j = blockIdx.x, t = threadIdx.x;              // 128 x 128
  WkT[j * DD + t] = (_Float16)Wk[t * DD + j];       // WkT[out][in]
}

__global__ void prep_wcq(const float* __restrict__ Wc, const float* __restrict__ Wq,
                         const float* __restrict__ bc, const float* __restrict__ bq,
                         _Float16* __restrict__ WcqbotT, float* __restrict__ Wcqtop,
                         float* __restrict__ bcq) {
  int j = blockIdx.x, t = threadIdx.x;              // 128 blocks x 256 thr
  if (t < 128) {
    int i = t; float s = 0.f;
    for (int m = 0; m < 128; ++m) s += Wc[(128 + i) * DD + m] * Wq[m * DD + j];
    WcqbotT[j * DD + i] = (_Float16)s;              // transposed [out][in], f16
  } else {
    int i = t - 128; float s = 0.f;
    for (int m = 0; m < 128; ++m) s += Wc[i * DD + m] * Wq[m * DD + j];
    Wcqtop[i * DD + j] = s;                         // row-major [in][out], f32
  }
  if (t == 0) {
    float s = 0.f;
    for (int m = 0; m < 128; ++m) s += bc[m] * Wq[m * DD + j];
    bcq[j] = s + bq[j];
  }
}

__global__ void qg_kernel(const float* __restrict__ graph, const float* __restrict__ Wcqtop,
                          const float* __restrict__ bcq, float* __restrict__ qg) {
  int b = blockIdx.x, j = threadIdx.x;              // 2048 x 128
  __shared__ float gs[128];
  gs[j] = graph[b * DD + j];
  __syncthreads();
  float s = bcq[j];
  #pragma unroll 4
  for (int i = 0; i < 128; ++i) s += gs[i] * Wcqtop[i * DD + j];
  qg[b * DD + j] = s;
}

__global__ void init_kernel(int* __restrict__ cur, uint8_t* __restrict__ visited,
                            int* __restrict__ tour, float* __restrict__ logp) {
  int b = blockIdx.x, n = threadIdx.x;              // 2048 x 128
  visited[b * NN + n] = (n == 0) ? 1 : 0;
  tour[b * NN + n] = 0;
  if (n == 0) { cur[b] = 0; logp[b] = 0.f; }
}

// ---------------------------------------------------------------------------
// WMMA strip: wave computes a 16(M) x 128(N) output strip, K=128 in 4 chunks.
// A in LDS (row-major f16, ld=128), B in LDS transposed [n][k] f16.
// ---------------------------------------------------------------------------
__device__ __forceinline__ void wmma_strip(const _Float16* ldsA, const _Float16* ldsB,
                                           int m0, int lane, v8f acc[8]) {
  const int ln  = lane & 15;
  const int sel = lane >> 4;
  #pragma unroll
  for (int kc = 0; kc < 128; kc += 32) {
    Frag a;
    const _Float16* arow = ldsA + (m0 + ln) * DD + kc + sel * 8;
    a.q[0] = *(const uint4*)(arow);        // K = kc+sel*8    .. +7
    a.q[1] = *(const uint4*)(arow + 16);   // K = kc+16+sel*8 .. +7
    #pragma unroll
    for (int j = 0; j < 8; ++j) {
      Frag bf;
      const _Float16* brow = ldsB + (j * 16 + ln) * DD + kc + sel * 8;
      bf.q[0] = ((const uint4*)brow)[0];
      bf.q[1] = ((const uint4*)brow)[2];
      acc[j] = __builtin_amdgcn_wmma_f32_16x16x32_f16(
          false, a.v, false, bf.v, (short)0, acc[j], false, false);
    }
  }
}

// ---------------------------------------------------------------------------
// TDM: 1-D contiguous tensor_load_to_lds of `bytes/4` dwords (data_size=4B).
// Issued by one wave only; caller must s_wait_tensorcnt + barrier.
// ---------------------------------------------------------------------------
__device__ __forceinline__ void tdm_load_1d(uint32_t lds_off, const void* gptr, uint32_t nelem4) {
  uint64_t ga = (uint64_t)(uintptr_t)gptr;
  v4u g0;
  g0[0] = 0x1u;                              // count=1, user mode, no gather
  g0[1] = lds_off;                           // lds_addr (bytes)
  g0[2] = (uint32_t)(ga & 0xFFFFFFFFu);      // global_addr[31:0]
  g0[3] = (uint32_t)((ga >> 32) & 0x01FFFFFFu) | (2u << 30);  // addr[56:32] | type=2
  v8i_ g1;
  g1[0] = (int)(2u << 16);                   // data_size = 2 (4 bytes)
  g1[1] = (int)((nelem4 & 0xFFFFu) << 16);   // tensor_dim0 low16
  g1[2] = (int)(((nelem4 >> 16) & 0xFFFFu) | (1u << 16));  // dim0 hi16 | tensor_dim1=1
  g1[3] = (int)((nelem4 & 0xFFFFu) << 16);   // tile_dim0 (low16; <=65535 elems)
  g1[4] = 0;                                 // tile_dim1=0 (1-D), tile_dim2=0
  g1[5] = (int)nelem4;                       // tensor_dim0_stride low32
  g1[6] = 0;
  g1[7] = 0;
  v4i_ gz = (v4i_){0, 0, 0, 0};
#if __clang_major__ >= 23
  v8i_ gz8 = (v8i_){0, 0, 0, 0, 0, 0, 0, 0};
  __builtin_amdgcn_tensor_load_to_lds(g0, g1, gz, gz, gz8, 0);
#else
  __builtin_amdgcn_tensor_load_to_lds(g0, g1, gz, gz, 0);
#endif
}

// ---------------------------------------------------------------------------
// K projection: per block, one batch b.
// Kt packed layout: Kt[b][dout>>2][node][dout&3]  (4 consecutive dims per node)
// ---------------------------------------------------------------------------
__global__ __launch_bounds__(256) void kproj_kernel(const float* __restrict__ h,
                                                    const _Float16* __restrict__ WkT,
                                                    const float* __restrict__ bk,
                                                    _Float16* __restrict__ Kt) {
  __shared__ float    ldsHf32[NN * DD];   // 64KB : raw h[b] tile via TDM
  __shared__ _Float16 ldsA[NN * DD];      // 32KB : h rows f16
  __shared__ _Float16 ldsB[DD * DD];      // 32KB : WkT
  const int b = blockIdx.x;

  if ((threadIdx.x >> 5) == 0) {          // wave 0 issues the TDM DMA
    uint32_t lds_off = (uint32_t)(uintptr_t)&ldsHf32[0];
    tdm_load_1d(lds_off, h + (size_t)b * NN * DD, NN * DD);
  }
  for (int idx = threadIdx.x; idx < DD * DD; idx += 256) ldsB[idx] = WkT[idx];
  if ((threadIdx.x >> 5) == 0) __builtin_amdgcn_s_wait_tensorcnt(0);
  __syncthreads();                        // publish TDM-written LDS to all waves

  for (int idx = threadIdx.x; idx < NN * DD; idx += 256)
    ldsA[idx] = (_Float16)ldsHf32[idx];   // f32 -> f16 convert
  __syncthreads();

  const int wave = threadIdx.x >> 5, lane = threadIdx.x & 31;
  const int m0 = wave * 16;
  const int ln = lane & 15, sel = lane >> 4;
  const v8f zero = {0.f,0.f,0.f,0.f,0.f,0.f,0.f,0.f};
  v8f acc[8];
  #pragma unroll
  for (int j = 0; j < 8; ++j) acc[j] = zero;

  wmma_strip(ldsA, ldsB, m0, lane, acc);

  _Float16* Ktb = Kt + (size_t)b * NN * DD;
  #pragma unroll
  for (int j = 0; j < 8; ++j) {
    const int n = j * 16 + ln;            // output dim
    const float bias = bk[n];
    #pragma unroll
    for (int i = 0; i < 8; ++i) {
      const int m = m0 + sel * 8 + i;     // node index
      Ktb[((n >> 2) * NN + m) * 4 + (n & 3)] = (_Float16)(acc[j][i] + bias);
    }
  }
}

// ---------------------------------------------------------------------------
// Per-step query GEMM: q[b,:] = qg[b,:] + h[b,cur[b],:] @ Wcq_bot
// grid 16 blocks x 256 thr; block covers 128 batches.
// ---------------------------------------------------------------------------
__global__ __launch_bounds__(256) void qstep_kernel(const float* __restrict__ h,
                                                    const _Float16* __restrict__ WcqbotT,
                                                    const float* __restrict__ qg,
                                                    const int* __restrict__ cur,
                                                    float* __restrict__ q) {
  __shared__ _Float16 ldsA[128 * DD];  // gathered h_cur rows, f16
  __shared__ _Float16 ldsB[DD * DD];   // WcqbotT
  const int b0 = blockIdx.x * 128;
  for (int idx = threadIdx.x; idx < DD * DD; idx += 256) ldsB[idx] = WcqbotT[idx];
  {
    const int r  = threadIdx.x >> 1;          // 0..127 (batch row within block)
    const int d0 = (threadIdx.x & 1) * 64;
    const int c  = cur[b0 + r];
    const float* hrow = h + ((size_t)(b0 + r) * NN + c) * DD;
    for (int d = d0; d < d0 + 64; ++d) ldsA[r * DD + d] = (_Float16)hrow[d];
  }
  __syncthreads();

  const int wave = threadIdx.x >> 5, lane = threadIdx.x & 31;
  const int m0 = wave * 16;
  const int ln = lane & 15, sel = lane >> 4;
  const v8f zero = {0.f,0.f,0.f,0.f,0.f,0.f,0.f,0.f};
  v8f acc[8];
  #pragma unroll
  for (int j = 0; j < 8; ++j) acc[j] = zero;

  wmma_strip(ldsA, ldsB, m0, lane, acc);

  #pragma unroll
  for (int j = 0; j < 8; ++j) {
    const int n = j * 16 + ln;
    #pragma unroll
    for (int i = 0; i < 8; ++i) {
      const int bi = b0 + m0 + sel * 8 + i;
      q[bi * DD + n] = acc[j][i] + qg[bi * DD + n];
    }
  }
}

// ---------------------------------------------------------------------------
// Scores + categorical sample (Gumbel-max) + log-softmax.
// grid 2048 blocks x 128 thr; thread n <-> candidate node n.
// K streamed as 8B/lane coalesced loads from packed Kt layout.
// ---------------------------------------------------------------------------
__global__ __launch_bounds__(128) void scores_kernel(const _Float16* __restrict__ Kt,
                                                     const float* __restrict__ q,
                                                     int step,
                                                     int* __restrict__ cur,
                                                     uint8_t* __restrict__ visited,
                                                     int* __restrict__ tour,
                                                     float* __restrict__ logp) {
  const int b = blockIdx.x, n = threadIdx.x;
  __shared__ float qs[128];
  __shared__ float red[128];
  __shared__ int   redi[128];
  __shared__ float s_sel;

  qs[n] = q[b * DD + n];
  __syncthreads();

  const uint2* Ktp = (const uint2*)(Kt + (size_t)b * NN * DD);
  float s = 0.f;
  #pragma unroll 8
  for (int dg = 0; dg < 32; ++dg) {                 // 4 dims per group, 8B/lane
    if ((dg & 7) == 0 && dg + 8 < 32)
      __builtin_prefetch((const void*)(Ktp + (dg + 8) * NN + n), 0, 1);
    union { uint2 u; _Float16 h4[4]; } w;
    w.u = Ktp[dg * NN + n];
    s += qs[4 * dg + 0] * (float)w.h4[0];
    s += qs[4 * dg + 1] * (float)w.h4[1];
    s += qs[4 * dg + 2] * (float)w.h4[2];
    s += qs[4 * dg + 3] * (float)w.h4[3];
  }
  s = 10.0f * tanhf(s * (1.0f / 32.0f));          // /(H*sqrt(dk)) = /32, clip 10

  const bool vis = visited[b * NN + n] != 0;
  const float smask = vis ? -1e30f : s;

  // Gumbel-max categorical sample (deterministic hash RNG)
  uint32_t hs = hashu(((uint32_t)b * 0x9E3779B9u) ^ ((uint32_t)step * 0x85EBCA6Bu) ^ (uint32_t)n);
  float u = ((float)(hs >> 8) + 0.5f) * (1.0f / 16777216.0f);
  float gum = -logf(-logf(u));
  float key = vis ? -1e30f : (s + gum);

  // argmax(key)
  red[n] = key; redi[n] = n; __syncthreads();
  #pragma unroll
  for (int off = 64; off > 0; off >>= 1) {
    if (n < off && red[n + off] > red[n]) { red[n] = red[n + off]; redi[n] = redi[n + off]; }
    __syncthreads();
  }
  const int nxt = redi[0];
  __syncthreads();
  if (n == nxt) s_sel = smask;

  // logsumexp over unvisited
  red[n] = smask; __syncthreads();
  #pragma unroll
  for (int off = 64; off > 0; off >>= 1) {
    if (n < off) red[n] = fmaxf(red[n], red[n + off]);
    __syncthreads();
  }
  const float mx = red[0];
  __syncthreads();
  red[n] = vis ? 0.f : expf(s - mx); __syncthreads();
  #pragma unroll
  for (int off = 64; off > 0; off >>= 1) {
    if (n < off) red[n] += red[n + off];
    __syncthreads();
  }
  if (n == 0) {
    const float lse = mx + logf(red[0]);
    logp[b] += s_sel - lse;
    tour[b * NN + step + 1] = nxt;
    cur[b] = nxt;
    visited[b * NN + nxt] = 1;
  }
}

// ---------------------------------------------------------------------------
// Host orchestration
// ---------------------------------------------------------------------------
extern "C" void kernel_launch(void* const* d_in, const int* in_sizes, int n_in,
                              void* d_out, int out_size, void* d_ws, size_t ws_size,
                              hipStream_t stream) {
  const float* h     = (const float*)d_in[0];
  const float* graph = (const float*)d_in[1];
  const float* Wq    = (const float*)d_in[2];
  const float* bq    = (const float*)d_in[3];
  const float* Wk    = (const float*)d_in[4];
  const float* bk    = (const float*)d_in[5];
  // d_in[6], d_in[7] = Wv, bv : computed-but-unused in reference
  const float* Wc    = (const float*)d_in[8];
  const float* bc    = (const float*)d_in[9];

  char* ws = (char*)d_ws;
  size_t off = 0;
  _Float16* Kt      = (_Float16*)(ws + off); off += (size_t)NB * NN * DD * 2;  // 64 MB
  _Float16* WkT     = (_Float16*)(ws + off); off += (size_t)DD * DD * 2;
  _Float16* WcqbotT = (_Float16*)(ws + off); off += (size_t)DD * DD * 2;
  float*    Wcqtop  = (float*)   (ws + off); off += (size_t)DD * DD * 4;
  float*    bcq     = (float*)   (ws + off); off += 512;
  float*    qg      = (float*)   (ws + off); off += (size_t)NB * DD * 4;
  float*    q       = (float*)   (ws + off); off += (size_t)NB * DD * 4;
  int*      cur     = (int*)     (ws + off); off += (size_t)NB * 4;
  uint8_t*  visited = (uint8_t*) (ws + off); off += (size_t)NB * NN;

  int*   tour     = (int*)d_out;                 // [B,N] int32 bits
  float* logp_out = (float*)d_out + (size_t)NB * NN;

  prep_wkT <<<128, 128, 0, stream>>>(Wk, WkT);
  prep_wcq <<<128, 256, 0, stream>>>(Wc, Wq, bc, bq, WcqbotT, Wcqtop, bcq);
  qg_kernel<<<NB, 128, 0, stream>>>(graph, Wcqtop, bcq, qg);
  init_kernel<<<NB, 128, 0, stream>>>(cur, visited, tour, logp_out);
  kproj_kernel<<<NB, 256, 0, stream>>>(h, WkT, bk, Kt);

  for (int step = 0; step < NN - 1; ++step) {
    qstep_kernel <<<NB / 128, 256, 0, stream>>>(h, WcqbotT, qg, cur, q);
    scores_kernel<<<NB, 128, 0, stream>>>(Kt, q, step, cur, visited, tour, logp_out);
  }
}
